// MultiHeadAttention_9775345566262
// MI455X (gfx1250) — compile-verified, matrix operand order checked by reading, not run on hardware
//
#include <hip/hip_runtime.h>

#define B_  16
#define S_  1024
#define D_  768
#define H_  12
#define DH_ 64

typedef __attribute__((ext_vector_type(16))) __bf16 bf16x16;
typedef __attribute__((ext_vector_type(8)))  float  f32x8;

union FragBF { unsigned int u[8]; bf16x16 v; };

#define WMMA_BF16(a, b, c) \
    __builtin_amdgcn_wmma_f32_16x16x32_bf16(false, (a), false, (b), (short)0, (c), false, false)

__device__ __forceinline__ unsigned short f32_to_bf16(float f) {
    unsigned int u = __float_as_uint(f);
    u += 0x7FFFu + ((u >> 16) & 1u);           // round-to-nearest-even
    return (unsigned short)(u >> 16);
}

// k-pair layout for 16-bit A/B fragments (ISA 7.12.2):
// vgpr r holds k = {base, base+1}, base = ((r&4)<<2) + ((r&3)<<1) + (half<<3)
__device__ __forceinline__ int pairk(int r, int h2) {
    return ((r & 4) << 2) + ((r & 3) << 1) + (h2 << 3);
}

// Fragment loader where consecutive-k elements are contiguous in memory:
// p = &M[lane_row_or_col][k_base]; each VGPR is one dword (merges to b128).
__device__ __forceinline__ FragBF load_frag(const unsigned short* p, int h2) {
    FragBF f;
#pragma unroll
    for (int r = 0; r < 8; ++r)
        f.u[r] = *(const unsigned int*)(p + pairk(r, h2));
    return f;
}

// Async global->LDS copy of 16 bytes per lane (GLOBAL_LOAD_ASYNC_TO_LDS_B128,
// tracked by ASYNCcnt). ldsoff = wave-relative LDS byte address (low 32 bits
// of a generic pointer to __shared__).
__device__ __forceinline__ void async_copy_b128(unsigned int ldsoff,
                                                const void* gaddr) {
    asm volatile("global_load_async_to_lds_b128 %0, %1, off"
                 :: "v"(ldsoff), "v"((unsigned long long)(uintptr_t)gaddr)
                 : "memory");
}

// ---------------------------------------------------------------- converts --
__global__ void cvt_bf16_v4(const float* __restrict__ s, unsigned short* __restrict__ d,
                            int n4) {  // n4 = n/4, n % 4 == 0
    int i = blockIdx.x * blockDim.x + threadIdx.x;
    int stride = gridDim.x * blockDim.x;
    for (; i < n4; i += stride) {
        float4 f = ((const float4*)s)[i];
        unsigned int lo = (unsigned int)f32_to_bf16(f.x) | ((unsigned int)f32_to_bf16(f.y) << 16);
        unsigned int hi = (unsigned int)f32_to_bf16(f.z) | ((unsigned int)f32_to_bf16(f.w) << 16);
        ((uint2*)d)[i] = make_uint2(lo, hi);
    }
}

// [K,E] fp32 -> [E,K] bf16 per matrix (blockIdx.z selects matrix/head)
__global__ void cvt_t_bf16(const float* __restrict__ s, unsigned short* __restrict__ d,
                           int K, int E) {
    const size_t mat = (size_t)blockIdx.z * K * E;
    const float* sm = s + mat;
    unsigned short* dm = d + mat;
    int n = K * E;
    int i = blockIdx.x * blockDim.x + threadIdx.x;
    int stride = gridDim.x * blockDim.x;
    for (; i < n; i += stride) {
        int k = i / E, e = i - k * E;
        dm[(size_t)e * K + k] = f32_to_bf16(sm[i]);
    }
}

// ---------------------------------------------------------------- QKV GEMM --
// grid (128, 12, 3), block 256. wave -> 16 rows x 64 cols (one full head);
// A-fragment reused across the 4 N-tiles. blockIdx.y = h, blockIdx.z = Q/K/V.
__global__ void __launch_bounds__(256) qkv_gemm(
    const unsigned short* __restrict__ Xb,
    const unsigned short* __restrict__ Wqt, const unsigned short* __restrict__ Wkt,
    const unsigned short* __restrict__ Wvt,
    const float* __restrict__ bq, const float* __restrict__ bk, const float* __restrict__ bv,
    unsigned short* __restrict__ Qb, unsigned short* __restrict__ Kb,
    unsigned short* __restrict__ Vb)
{
    const int lane = threadIdx.x & 31, wv = threadIdx.x >> 5;
    const int h2 = lane >> 4, ln16 = lane & 15;
    const int mtile = blockIdx.x * 8 + wv;                 // 0..1023
    const int h = blockIdx.y;
    const int z = blockIdx.z;
    const unsigned short* Wt  = (z == 0) ? Wqt : (z == 1) ? Wkt : Wvt;
    const float*          bia = (z == 0) ? bq  : (z == 1) ? bk  : bv;

    const unsigned short* arow = Xb + (size_t)(mtile * 16 + ln16) * D_;
    const unsigned short* w0 = Wt + ((size_t)h * DH_ +  0 + ln16) * D_;
    const unsigned short* w1 = Wt + ((size_t)h * DH_ + 16 + ln16) * D_;
    const unsigned short* w2 = Wt + ((size_t)h * DH_ + 32 + ln16) * D_;
    const unsigned short* w3 = Wt + ((size_t)h * DH_ + 48 + ln16) * D_;

    f32x8 a0 = {0.f,0.f,0.f,0.f,0.f,0.f,0.f,0.f};
    f32x8 a1 = a0, a2 = a0, a3 = a0;
    for (int kb = 0; kb < D_; kb += 32) {
        FragBF a = load_frag(arow + kb, h2);
        FragBF b0 = load_frag(w0 + kb, h2);
        FragBF b1 = load_frag(w1 + kb, h2);
        FragBF b2 = load_frag(w2 + kb, h2);
        FragBF b3 = load_frag(w3 + kb, h2);
        a0 = WMMA_BF16(a.v, b0.v, a0);
        a1 = WMMA_BF16(a.v, b1.v, a1);
        a2 = WMMA_BF16(a.v, b2.v, a2);
        a3 = WMMA_BF16(a.v, b3.v, a3);
    }
#pragma unroll
    for (int nt = 0; nt < 4; ++nt) {
        const f32x8& acc = (nt == 0) ? a0 : (nt == 1) ? a1 : (nt == 2) ? a2 : a3;
        const float bval = bia[h * DH_ + nt * 16 + ln16];
#pragma unroll
        for (int r = 0; r < 8; ++r) {
            int mrow = mtile * 16 + r + (h2 << 3);
            int bb = mrow >> 10, s = mrow & (S_ - 1);
            unsigned short val = f32_to_bf16(acc[r] + bval);
            if (z == 2) {  // V transposed: [b,h,dh,s]
                Vb[(((size_t)bb * H_ + h) * DH_ + nt * 16 + ln16) * S_ + s] = val;
            } else {
                unsigned short* out = (z == 0) ? Qb : Kb;
                out[(((size_t)bb * H_ + h) * S_ + s) * DH_ + nt * 16 + ln16] = val;
            }
        }
    }
}

// ---------------------------------------------------- fused flash attention --
// grid (B*H = 192, 8), block 256 (8 waves, one 16-row Q tile each, same b,h).
// K/V tiles (32 keys) staged in double-buffered LDS via async global->LDS DMA.
__global__ void __launch_bounds__(256) attn_kernel(
    const unsigned short* __restrict__ Q, const unsigned short* __restrict__ K,
    const unsigned short* __restrict__ Vt, unsigned short* __restrict__ msa)
{
    __shared__ __align__(16) unsigned short Klds[2][32 * DH_];  // [key][dh]  4KB x2
    __shared__ __align__(16) unsigned short Vlds[2][DH_ * 32];  // [dh][key]  4KB x2
    __shared__ __align__(16) unsigned short Plds[8 * 512];      // 16x32 per wave

    const int t = threadIdx.x;
    const int lane = t & 31, wv = t >> 5;
    const int h2 = lane >> 4, ln16 = lane & 15;
    const int bh = blockIdx.x;
    const int b = bh / H_, h = bh % H_;
    const int qtile = blockIdx.y * 8 + wv;                 // 0..63

    const unsigned short* Qh = Q  + (size_t)bh * S_ * DH_;
    const unsigned short* Kh = K  + (size_t)bh * S_ * DH_;
    const unsigned short* Vh = Vt + (size_t)bh * DH_ * S_;

    const unsigned short* qrow = Qh + (size_t)(qtile * 16 + ln16) * DH_;
    FragBF qa0 = load_frag(qrow, h2);                      // k = 0..31  (Dh)
    FragBF qa1 = load_frag(qrow + 32, h2);                 // k = 32..63 (Dh)

    f32x8 o0 = {0.f,0.f,0.f,0.f,0.f,0.f,0.f,0.f};
    f32x8 o1 = o0, o2 = o0, o3 = o0;
    float mrun[8], lrun[8];
#pragma unroll
    for (int r = 0; r < 8; ++r) { mrun[r] = -1e30f; lrun[r] = 0.f; }

    unsigned short* pl = Plds + wv * 512;

    // per-thread LDS byte addresses for the async tile copies
    const unsigned int kdst[2] = {
        (unsigned int)(uintptr_t)(Klds[0] + (size_t)t * 8),
        (unsigned int)(uintptr_t)(Klds[1] + (size_t)t * 8) };
    const unsigned int vdst[2] = {
        (unsigned int)(uintptr_t)(Vlds[0] + (t >> 2) * 32 + (t & 3) * 8),
        (unsigned int)(uintptr_t)(Vlds[1] + (t >> 2) * 32 + (t & 3) * 8) };

    for (int j0 = 0; j0 < S_; j0 += 32) {
        const int p = (j0 >> 5) & 1;
        // ---- async stage of K/V tiles into LDS (4KB each, ASYNCcnt) ------
        // K tile: rows j0..j0+31 contiguous [32][64]; V tile: 64 dh-rows of 32.
        async_copy_b128(kdst[p], Kh + (size_t)j0 * DH_ + (size_t)t * 8);
        async_copy_b128(vdst[p], Vh + (size_t)(t >> 2) * S_ + j0 + (t & 3) * 8);
        if (j0 + 32 < S_) {   // prefetch next tiles -> global_prefetch_b8
            __builtin_prefetch(Kh + (size_t)(j0 + 32) * DH_ + (size_t)t * 8, 0, 1);
            __builtin_prefetch(Vh + (size_t)(t >> 2) * S_ + j0 + 32 + (t & 3) * 8, 0, 1);
        }
        asm volatile("s_wait_asynccnt 0" ::: "memory");    // tile resident
        __syncthreads();

        // ---- scores: S[16x32] = Q[16x64] . K^T (from LDS) ----------------
        const unsigned short* k0 = Klds[p] + (size_t)ln16 * DH_;
        const unsigned short* k1 = Klds[p] + (size_t)(16 + ln16) * DH_;
        FragBF kb00 = load_frag(k0,      h2), kb01 = load_frag(k0 + 32, h2);
        FragBF kb10 = load_frag(k1,      h2), kb11 = load_frag(k1 + 32, h2);
        f32x8 s0 = {0.f,0.f,0.f,0.f,0.f,0.f,0.f,0.f};
        f32x8 s1 = s0;
        s0 = WMMA_BF16(qa0.v, kb00.v, s0);
        s0 = WMMA_BF16(qa1.v, kb01.v, s0);
        s1 = WMMA_BF16(qa0.v, kb10.v, s1);
        s1 = WMMA_BF16(qa1.v, kb11.v, s1);

        // ---- online softmax over the 32 new columns ----------------------
#pragma unroll
        for (int r = 0; r < 8; ++r) {
            float a = s0[r] * 0.125f;                      // 1/sqrt(64)
            float c = s1[r] * 0.125f;
            float cand = fmaxf(a, c);
            cand = fmaxf(cand, __shfl_xor(cand, 1, 32));
            cand = fmaxf(cand, __shfl_xor(cand, 2, 32));
            cand = fmaxf(cand, __shfl_xor(cand, 4, 32));
            cand = fmaxf(cand, __shfl_xor(cand, 8, 32));
            float mnew = fmaxf(mrun[r], cand);
            float fs = __expf(mrun[r] - mnew);
            float p0 = __expf(a - mnew);
            float p1 = __expf(c - mnew);
            float rs = p0 + p1;
            rs += __shfl_xor(rs, 1, 32);
            rs += __shfl_xor(rs, 2, 32);
            rs += __shfl_xor(rs, 4, 32);
            rs += __shfl_xor(rs, 8, 32);
            lrun[r] = lrun[r] * fs + rs;
            mrun[r] = mnew;
            o0[r] *= fs; o1[r] *= fs; o2[r] *= fs; o3[r] *= fs;
            int row = r + (h2 << 3);                       // C-layout row
            pl[row * 32 + ln16]      = f32_to_bf16(p0);
            pl[row * 32 + 16 + ln16] = f32_to_bf16(p1);
        }
        asm volatile("s_wait_dscnt 0" ::: "memory");       // intra-wave LDS RAW

        // re-read P in A-fragment layout (16x32)
        FragBF pa;
#pragma unroll
        for (int r = 0; r < 8; ++r)
            pa.u[r] = *(const unsigned int*)(pl + ln16 * 32 + pairk(r, h2));

        // ---- O[16x64] += P[16x32] . V[32x64] (V from LDS) ----------------
        FragBF vb0 = load_frag(Vlds[p] + (size_t)(ln16)      * 32, h2);
        FragBF vb1 = load_frag(Vlds[p] + (size_t)(16 + ln16) * 32, h2);
        FragBF vb2 = load_frag(Vlds[p] + (size_t)(32 + ln16) * 32, h2);
        FragBF vb3 = load_frag(Vlds[p] + (size_t)(48 + ln16) * 32, h2);
        o0 = WMMA_BF16(pa.v, vb0.v, o0);
        o1 = WMMA_BF16(pa.v, vb1.v, o1);
        o2 = WMMA_BF16(pa.v, vb2.v, o2);
        o3 = WMMA_BF16(pa.v, vb3.v, o3);
    }

    // ---- normalize and write concat-head bf16 output ----------------------
#pragma unroll
    for (int r = 0; r < 8; ++r) {
        float inv = 1.0f / lrun[r];
        int mrow = qtile * 16 + r + (h2 << 3);
        size_t base = ((size_t)b * S_ + mrow) * D_ + h * DH_ + ln16;
        msa[base + 0]  = f32_to_bf16(o0[r] * inv);
        msa[base + 16] = f32_to_bf16(o1[r] * inv);
        msa[base + 32] = f32_to_bf16(o2[r] * inv);
        msa[base + 48] = f32_to_bf16(o3[r] * inv);
    }
}

// ------------------------------------------------------- output projection --
// grid (128, 12), block 256. wave -> 16x64 strip, A-fragment reused 4x.
__global__ void __launch_bounds__(256) out_gemm(
    const unsigned short* __restrict__ Ab, const unsigned short* __restrict__ Wot,
    const float* __restrict__ bo, float* __restrict__ out)
{
    const int lane = threadIdx.x & 31, wv = threadIdx.x >> 5;
    const int h2 = lane >> 4, ln16 = lane & 15;
    const int mtile = blockIdx.x * 8 + wv;                 // 0..1023
    const int nbase = blockIdx.y * 64;

    const unsigned short* arow = Ab  + (size_t)(mtile * 16 + ln16) * D_;
    const unsigned short* w0 = Wot + (size_t)(nbase +  0 + ln16) * D_;
    const unsigned short* w1 = Wot + (size_t)(nbase + 16 + ln16) * D_;
    const unsigned short* w2 = Wot + (size_t)(nbase + 32 + ln16) * D_;
    const unsigned short* w3 = Wot + (size_t)(nbase + 48 + ln16) * D_;

    f32x8 a0 = {0.f,0.f,0.f,0.f,0.f,0.f,0.f,0.f};
    f32x8 a1 = a0, a2 = a0, a3 = a0;
    for (int kb = 0; kb < D_; kb += 32) {
        FragBF a = load_frag(arow + kb, h2);
        FragBF b0 = load_frag(w0 + kb, h2);
        FragBF b1 = load_frag(w1 + kb, h2);
        FragBF b2 = load_frag(w2 + kb, h2);
        FragBF b3 = load_frag(w3 + kb, h2);
        a0 = WMMA_BF16(a.v, b0.v, a0);
        a1 = WMMA_BF16(a.v, b1.v, a1);
        a2 = WMMA_BF16(a.v, b2.v, a2);
        a3 = WMMA_BF16(a.v, b3.v, a3);
    }
#pragma unroll
    for (int nt = 0; nt < 4; ++nt) {
        const f32x8& acc = (nt == 0) ? a0 : (nt == 1) ? a1 : (nt == 2) ? a2 : a3;
        const int ncol = nbase + nt * 16 + ln16;
        const float bias = bo[ncol];
#pragma unroll
        for (int r = 0; r < 8; ++r) {
            int mrow = mtile * 16 + r + (h2 << 3);
            out[(size_t)mrow * D_ + ncol] = acc[r] + bias;
        }
    }
}

// ---------------------------------------------------------------------------
extern "C" void kernel_launch(void* const* d_in, const int* in_sizes, int n_in,
                              void* d_out, int out_size, void* d_ws, size_t ws_size,
                              hipStream_t stream) {
    (void)in_sizes; (void)n_in; (void)out_size; (void)ws_size;
    const float* X  = (const float*)d_in[0];
    const float* Wq = (const float*)d_in[1];
    const float* bq = (const float*)d_in[2];
    const float* Wk = (const float*)d_in[3];
    const float* bk = (const float*)d_in[4];
    const float* Wv = (const float*)d_in[5];
    const float* bv = (const float*)d_in[6];
    const float* Wo = (const float*)d_in[7];
    const float* bo = (const float*)d_in[8];
    float* out = (float*)d_out;

    const size_t nX  = (size_t)B_ * S_ * D_;       // 12,582,912
    const size_t nW  = (size_t)H_ * D_ * DH_;      //    589,824
    const size_t nWo = (size_t)D_ * D_;            //    589,824

    unsigned short* w = (unsigned short*)d_ws;
    size_t off = 0;
    unsigned short* Xb   = w + off; off += nX;
    unsigned short* Wqt  = w + off; off += nW;
    unsigned short* Wkt  = w + off; off += nW;
    unsigned short* Wvt  = w + off; off += nW;
    unsigned short* Wot  = w + off; off += nWo;
    unsigned short* Qb   = w + off; off += nX;     // [B,H,S,Dh]
    unsigned short* Kb   = w + off; off += nX;     // [B,H,S,Dh]
    unsigned short* Vb   = w + off; off += nX;     // [B,H,Dh,S] (transposed)
    unsigned short* msab = w + off; off += nX;     // [B,S,D]

    cvt_bf16_v4<<<2048, 256, 0, stream>>>(X, Xb, (int)(nX / 4));
    cvt_t_bf16<<<dim3(48, 1, H_), 256, 0, stream>>>(Wq, Wqt, D_, DH_);
    cvt_t_bf16<<<dim3(48, 1, H_), 256, 0, stream>>>(Wk, Wkt, D_, DH_);
    cvt_t_bf16<<<dim3(48, 1, H_), 256, 0, stream>>>(Wv, Wvt, D_, DH_);
    cvt_t_bf16<<<dim3(256, 1, 1), 256, 0, stream>>>(Wo, Wot, D_, D_);

    qkv_gemm<<<dim3(128, H_, 3), 256, 0, stream>>>(Xb, Wqt, Wkt, Wvt,
                                                   bq, bk, bv, Qb, Kb, Vb);
    attn_kernel<<<dim3(B_ * H_, 8), 256, 0, stream>>>(Qb, Kb, Vb, msab);
    out_gemm<<<dim3(128, H_), 256, 0, stream>>>(msab, Wot, bo, out);
}
